// ObjectStateUpdater_90022514524502
// MI455X (gfx1250) — compile-verified, matrix operand order checked by reading, not run on hardware
//
#include <hip/hip_runtime.h>
#include <hip/hip_bf16.h>

typedef __attribute__((ext_vector_type(16))) _Float16 v16h;
typedef __attribute__((ext_vector_type(8)))  _Float16 v8h;
typedef __attribute__((ext_vector_type(8)))  float    v8f;

// ---------------------------------------------------------------------------
// WMMA helpers (CDNA5 wave32, 16x16x32 f16 -> f32 accumulate)
// ---------------------------------------------------------------------------
__device__ __forceinline__ v8f wmma16(v16h a, v16h b, v8f c) {
  // 8 args: (neg_a, A, neg_b, B, c_mod, C, reuse_a, reuse_b)
  return __builtin_amdgcn_wmma_f32_16x16x32_f16(false, a, false, b,
                                                (short)0, c, false, false);
}

// A/B fragment layout (ISA 7.12.2, 16-bit 16x32):
//   lane L: row/col = L%16, group g = L/16
//   halves j=0..7  -> K = kbase + 8*g + j
//   halves j=8..15 -> K = kbase + 16 + 8*g + (j-8)
// p points at (row_base_ptr + kbase)
__device__ __forceinline__ v16h frag_from_f32(const float* p, int g) {
  const float4* q0 = (const float4*)(p + 8 * g);
  const float4* q1 = (const float4*)(p + 16 + 8 * g);
  float4 f0 = q0[0], f1 = q0[1], f2 = q1[0], f3 = q1[1];
  v16h a;
  a[0]=(_Float16)f0.x;  a[1]=(_Float16)f0.y;  a[2]=(_Float16)f0.z;  a[3]=(_Float16)f0.w;
  a[4]=(_Float16)f1.x;  a[5]=(_Float16)f1.y;  a[6]=(_Float16)f1.z;  a[7]=(_Float16)f1.w;
  a[8]=(_Float16)f2.x;  a[9]=(_Float16)f2.y;  a[10]=(_Float16)f2.z; a[11]=(_Float16)f2.w;
  a[12]=(_Float16)f3.x; a[13]=(_Float16)f3.y; a[14]=(_Float16)f3.z; a[15]=(_Float16)f3.w;
  return a;
}

__device__ __forceinline__ v16h frag_from_f16(const _Float16* p, int g) {
  v8h h0 = *(const v8h*)(p + 8 * g);
  v8h h1 = *(const v8h*)(p + 16 + 8 * g);
  v16h a;
#pragma unroll
  for (int j = 0; j < 8; ++j) { a[j] = h0[j]; a[8 + j] = h1[j]; }
  return a;
}

// ---------------------------------------------------------------------------
// Generic strip GEMM: Y[N,128] = act( X[N,Kx] @ W[128,Kx]^T + bias )
// X may be a concat of two f32 sources (K0 | Kx-K0), or a single f16 source.
// One wave computes a 16-row x 128-col strip.
// ---------------------------------------------------------------------------
template <int IN_HALF, int OUT_HALF, int RELU>
__global__ void wmma_gemm_bias(const void* __restrict__ X0v,
                               const void* __restrict__ X1v,
                               int K0, int Kx,
                               const _Float16* __restrict__ W,
                               const float* __restrict__ bias,
                               void* __restrict__ Yv, int Nrows) {
  int wave = blockIdx.x * (blockDim.x >> 5) + (threadIdx.x >> 5);
  int rowbase = wave * 16;
  if (rowbase >= Nrows) return;           // wave-uniform: EXEC stays all-1s
  int lane = threadIdx.x & 31;
  int m = lane & 15;
  int g = lane >> 4;

  const v8f vz = {0.f, 0.f, 0.f, 0.f, 0.f, 0.f, 0.f, 0.f};
  v8f acc[8];
#pragma unroll
  for (int t = 0; t < 8; ++t) acc[t] = vz;

  for (int k = 0; k < Kx; k += 32) {
    v16h a;
    if (IN_HALF) {
      const _Float16* X = (const _Float16*)X0v;
      a = frag_from_f16(X + (size_t)(rowbase + m) * Kx + k, g);
    } else {
      const float* X; int kk, ld;
      if (k < K0) { X = (const float*)X0v; kk = k;       ld = K0; }
      else        { X = (const float*)X1v; kk = k - K0;  ld = Kx - K0; }
      a = frag_from_f32(X + (size_t)(rowbase + m) * ld + kk, g);
    }
#pragma unroll
    for (int t = 0; t < 8; ++t) {
      v16h b = frag_from_f16(W + (size_t)(t * 16 + m) * Kx + k, g);
      acc[t] = wmma16(a, b, acc[t]);
    }
  }

#pragma unroll
  for (int t = 0; t < 8; ++t) {
    float bv = bias[t * 16 + m];          // col = t*16 + lane%16
#pragma unroll
    for (int r = 0; r < 8; ++r) {
      int row = rowbase + r + 8 * g;
      float v = acc[t][r] + bv;
      if (RELU) v = v > 0.f ? v : 0.f;
      size_t o = (size_t)row * 128 + t * 16 + m;
      if (OUT_HALF) ((_Float16*)Yv)[o] = (_Float16)v;
      else          ((float*)Yv)[o]    = v;
    }
  }
}

// ---------------------------------------------------------------------------
// Fused GRU cell: gi = TE @ Wih^T + bih, gh = H @ Whh^T + bhh, gate math,
// out = (1-z)*n + z*h. One wave -> 16 rows x 16 output dims (blockIdx.y tile).
// ---------------------------------------------------------------------------
__global__ void gru_wmma_kernel(const float* __restrict__ TE,
                                const float* __restrict__ H,
                                const _Float16* __restrict__ Wih,
                                const _Float16* __restrict__ Whh,
                                const float* __restrict__ bih,
                                const float* __restrict__ bhh,
                                float* __restrict__ out, int Nrows) {
  int wave = blockIdx.x * (blockDim.x >> 5) + (threadIdx.x >> 5);
  int rowbase = wave * 16;
  if (rowbase >= Nrows) return;
  int lane = threadIdx.x & 31;
  int m = lane & 15;
  int g = lane >> 4;
  int dbase = blockIdx.y * 16;
  int d = dbase + m;

  const v8f vz = {0.f, 0.f, 0.f, 0.f, 0.f, 0.f, 0.f, 0.f};
  v8f air = vz, aiz = vz, ain = vz, ahr = vz, ahz = vz, ahn = vz;

#pragma unroll
  for (int k = 0; k < 128; k += 32) {
    v16h a_te = frag_from_f32(TE + (size_t)(rowbase + m) * 128 + k, g);
    v16h a_h  = frag_from_f32(H  + (size_t)(rowbase + m) * 128 + k, g);
    v16h b;
    b = frag_from_f16(Wih + (size_t)(dbase + m) * 128 + k, g);        air = wmma16(a_te, b, air);
    b = frag_from_f16(Wih + (size_t)(128 + dbase + m) * 128 + k, g);  aiz = wmma16(a_te, b, aiz);
    b = frag_from_f16(Wih + (size_t)(256 + dbase + m) * 128 + k, g);  ain = wmma16(a_te, b, ain);
    b = frag_from_f16(Whh + (size_t)(dbase + m) * 128 + k, g);        ahr = wmma16(a_h, b, ahr);
    b = frag_from_f16(Whh + (size_t)(128 + dbase + m) * 128 + k, g);  ahz = wmma16(a_h, b, ahz);
    b = frag_from_f16(Whh + (size_t)(256 + dbase + m) * 128 + k, g);  ahn = wmma16(a_h, b, ahn);
  }

  float bir = bih[d], biz = bih[128 + d], bin_v = bih[256 + d];
  float bhr = bhh[d], bhz = bhh[128 + d], bhn   = bhh[256 + d];

#pragma unroll
  for (int r = 0; r < 8; ++r) {
    int row = rowbase + r + 8 * g;
    float rg = 1.f / (1.f + __expf(-((air[r] + bir) + (ahr[r] + bhr))));
    float zg = 1.f / (1.f + __expf(-((aiz[r] + biz) + (ahz[r] + bhz))));
    float ng = tanhf((ain[r] + bin_v) + rg * (ahn[r] + bhn));
    float h  = H[(size_t)row * 128 + d];
    out[(size_t)row * 128 + d] = (1.f - zg) * ng + zg * h;
  }
}

// ---------------------------------------------------------------------------
// Edge gather/scatter-add: one wave per edge, lane -> 4 features (float4).
// dst[sidx[e]][:] += src[gidx[e]][:]; optional per-segment count.
// ---------------------------------------------------------------------------
__global__ void edge_scatter_kernel(const float4* __restrict__ src,
                                    const int* __restrict__ gidx,
                                    const int* __restrict__ sidx,
                                    float* __restrict__ dst,
                                    float* __restrict__ cnt, int n_edge) {
  int wave = blockIdx.x * (blockDim.x >> 5) + (threadIdx.x >> 5);
  if (wave >= n_edge) return;
  int lane = threadIdx.x & 31;
  int gi = gidx[wave];
  int si = sidx[wave];
  float4 v = src[(size_t)gi * 32 + lane];
  float* d = dst + (size_t)si * 128 + lane * 4;
  atomicAdd(d + 0, v.x);
  atomicAdd(d + 1, v.y);
  atomicAdd(d + 2, v.z);
  atomicAdd(d + 3, v.w);
  if (cnt != nullptr && lane == 0) atomicAdd(cnt + si, 1.0f);
}

__global__ void mean_kernel(float4* __restrict__ c, const float* __restrict__ cnt, int n4) {
  int i = blockIdx.x * blockDim.x + threadIdx.x;
  if (i >= n4) return;
  int row = i >> 5;                       // 32 float4 per 128-wide row
  float inv = 1.f / fmaxf(cnt[row], 1.f);
  float4 v = c[i];
  v.x *= inv; v.y *= inv; v.z *= inv; v.w *= inv;
  c[i] = v;
}

__global__ void zero4_kernel(float4* __restrict__ p, int n4) {
  int i = blockIdx.x * blockDim.x + threadIdx.x;
  if (i < n4) { float4 z = {0.f, 0.f, 0.f, 0.f}; p[i] = z; }
}

__global__ void f32_to_f16_kernel(const float* __restrict__ in, _Float16* __restrict__ out, int n) {
  int i = blockIdx.x * blockDim.x + threadIdx.x;
  if (i < n) out[i] = (_Float16)in[i];
}

// ---------------------------------------------------------------------------
// Orchestration
// ---------------------------------------------------------------------------
extern "C" void kernel_launch(void* const* d_in, const int* in_sizes, int n_in,
                              void* d_out, int out_size, void* d_ws, size_t ws_size,
                              hipStream_t stream) {
  (void)in_sizes; (void)n_in; (void)out_size; (void)ws_size;
  const float* event_X    = (const float*)d_in[0];
  const float* object_X   = (const float*)d_in[1];
  const int*   event_idx  = (const int*)d_in[2];
  const int*   object_idx = (const int*)d_in[3];
  const float* Wa  = (const float*)d_in[4];
  const float* ba  = (const float*)d_in[5];
  const float* W1  = (const float*)d_in[6];
  const float* b1  = (const float*)d_in[7];
  const float* W2  = (const float*)d_in[8];
  const float* b2  = (const float*)d_in[9];
  const float* Wih = (const float*)d_in[10];
  const float* Whh = (const float*)d_in[11];
  const float* bih = (const float*)d_in[12];
  const float* bhh = (const float*)d_in[13];

  const int N_OBJ = 100000, N_EVT = 200000, N_EDGE = 1000000;

  // Workspace layout (~155 MB):
  //  [0]            hproj / h_mid(f16) / total_effect : N_OBJ*128 f32 (51.2 MB)
  //  [+51.2MB]      c_sum -> c_atm : N_EVT*128 f32 (102.4 MB)
  //  [+153.6MB]     cnt : N_EVT f32
  //  [+154.4MB]     f16 weight copies (~320 KB)
  float*    hproj = (float*)d_ws;
  float*    csum  = hproj + (size_t)N_OBJ * 128;
  float*    cnt   = csum + (size_t)N_EVT * 128;
  _Float16* Wa_h  = (_Float16*)(cnt + N_EVT);
  _Float16* W1_h  = Wa_h + 128 * 128;
  _Float16* W2_h  = W1_h + 128 * 256;
  _Float16* Wih_h = W2_h + 128 * 128;
  _Float16* Whh_h = Wih_h + 384 * 128;
  _Float16* h_mid = (_Float16*)hproj;     // reuse after scatter-1 consumes hproj

  float* new_obj   = (float*)d_out;
  float* h_impulse = (float*)d_out + (size_t)N_OBJ * 128;

  // 1) f32 -> f16 weight copies
  f32_to_f16_kernel<<<(128 * 128 + 255) / 256, 256, 0, stream>>>(Wa,  Wa_h,  128 * 128);
  f32_to_f16_kernel<<<(128 * 256 + 255) / 256, 256, 0, stream>>>(W1,  W1_h,  128 * 256);
  f32_to_f16_kernel<<<(128 * 128 + 255) / 256, 256, 0, stream>>>(W2,  W2_h,  128 * 128);
  f32_to_f16_kernel<<<(384 * 128 + 255) / 256, 256, 0, stream>>>(Wih, Wih_h, 384 * 128);
  f32_to_f16_kernel<<<(384 * 128 + 255) / 256, 256, 0, stream>>>(Whh, Whh_h, 384 * 128);

  // 2) zero c_sum + cnt (contiguous)
  {
    int n4 = (N_EVT * 128 + N_EVT) / 4;   // 6,450,000
    zero4_kernel<<<(n4 + 255) / 256, 256, 0, stream>>>((float4*)csum, n4);
  }

  // 3) hproj = object_X @ Wa^T + ba
  {
    int waves = N_OBJ / 16, blocks = (waves + 3) / 4;
    wmma_gemm_bias<0, 0, 0><<<blocks, 128, 0, stream>>>(
        object_X, nullptr, 128, 128, Wa_h, ba, hproj, N_OBJ);
  }

  // 4) edge scatter-mean accumulate: c_sum[evt] += hproj[obj], cnt[evt] += 1
  edge_scatter_kernel<<<N_EDGE / 8, 256, 0, stream>>>(
      (const float4*)hproj, object_idx, event_idx, csum, cnt, N_EDGE);

  // 5) c_atm = c_sum / max(cnt, 1)
  {
    int n4 = N_EVT * 32;
    mean_kernel<<<(n4 + 255) / 256, 256, 0, stream>>>((float4*)csum, cnt, n4);
  }

  // 6) h_mid = relu([event_X | c_atm] @ W1^T + b1)   (f16 out, reuses hproj buf)
  {
    int waves = N_EVT / 16, blocks = (waves + 3) / 4;
    wmma_gemm_bias<0, 1, 1><<<blocks, 128, 0, stream>>>(
        event_X, csum, 128, 256, W1_h, b1, h_mid, N_EVT);
    // 6b) h_impulse = h_mid @ W2^T + b2  (written straight into d_out)
    wmma_gemm_bias<1, 0, 0><<<blocks, 128, 0, stream>>>(
        h_mid, nullptr, 128, 128, W2_h, b2, h_impulse, N_EVT);
  }

  // 7) zero total_effect (aliases hproj/h_mid buffer, both dead now)
  {
    int n4 = N_OBJ * 32;
    zero4_kernel<<<(n4 + 255) / 256, 256, 0, stream>>>((float4*)hproj, n4);
  }

  // 8) total_effect[obj] += h_impulse[evt] over edges
  edge_scatter_kernel<<<N_EDGE / 8, 256, 0, stream>>>(
      (const float4*)h_impulse, event_idx, object_idx, hproj, nullptr, N_EDGE);

  // 9) fused GRU cell -> new_object_X
  {
    int waves = N_OBJ / 16;
    dim3 grid((waves + 3) / 4, 8);
    gru_wmma_kernel<<<grid, 128, 0, stream>>>(
        hproj, object_X, Wih_h, Whh_h, bih, bhh, new_obj, N_OBJ);
  }
}
